// Head_3178275799485
// MI455X (gfx1250) — compile-verified, matrix operand order checked by reading, not run on hardware
//
#include <hip/hip_runtime.h>
#include <hip/hip_bf16.h>

typedef __bf16 bf16;
typedef __bf16 v16bf __attribute__((ext_vector_type(16)));
typedef __bf16 v8bf  __attribute__((ext_vector_type(8)));
typedef float  v8f   __attribute__((ext_vector_type(8)));

#define BB 8
#define TT 2048
#define CC 1024
#define HD 64

// LDS row paddings (elements). Byte strides: K 144B (36 dw), V/P 80B (20 dw).
// 36L mod 64 and 20L mod 64 are distinct for L=0..15 -> conflict-free
// 16-lane strided ds_load_b128; both are multiples of 16B for alignment.
#define KPAD 72
#define VPAD 40
#define PPAD 40

__device__ __forceinline__ v8f vzero() {
    v8f z;
#pragma unroll
    for (int i = 0; i < 8; ++i) z[i] = 0.0f;
    return z;
}

__device__ __forceinline__ v8f wmma_bf16(v16bf a, v16bf b, v8f c) {
    // D = A(16x32 bf16) * B(32x16 bf16) + C(16x16 f32)
    return __builtin_amdgcn_wmma_f32_16x16x32_bf16(
        false, a, false, b, (short)0, c, false, false);
}

// Load a 16x32 bf16 A-operand (or N-major B-operand) tile from row-major
// memory (global or LDS). Per-lane (wave32): row = lane&15; lanes 0-15 take
// K {0..7,16..23}, lanes 16-31 take K {8..15,24..31}. Two 16B loads per lane.
__device__ __forceinline__ v16bf load_ab16(const bf16* __restrict__ base,
                                           long stride, int lane) {
    int row  = lane & 15;
    int koff = (lane & 16) ? 8 : 0;
    const bf16* p = base + (long)row * stride + koff;
    union { v16bf v; v8bf h[2]; } u;
    u.h[0] = *(const v8bf*)(p);
    u.h[1] = *(const v8bf*)(p + 16);
    return u.v;
}

// CDNA5 async copy: global -> LDS, 16 bytes per lane, tracked by ASYNCcnt.
__device__ __forceinline__ void async_b128(const void* lds_dst, const void* src) {
    uint32_t loff = (uint32_t)(uintptr_t)lds_dst;   // flat-LDS low 32 bits = DS offset
    asm volatile("global_load_async_to_lds_b128 %0, %1, off"
                 :: "v"(loff), "v"(src) : "memory");
}
__device__ __forceinline__ void wait_async0() {
    asm volatile("s_wait_asynccnt 0x0" ::: "memory");
}

// ---------------- Kernel 0a: x (f32) -> xb (bf16), row-major ----------------
__global__ void cvt_x(const float* __restrict__ x, bf16* __restrict__ xb, int n8) {
    int i = blockIdx.x * blockDim.x + threadIdx.x;
    if (i >= n8) return;
    const float4* p = (const float4*)x + (long)i * 2;
    float4 a = p[0], b = p[1];
    v8bf o;
    o[0] = (bf16)a.x; o[1] = (bf16)a.y; o[2] = (bf16)a.z; o[3] = (bf16)a.w;
    o[4] = (bf16)b.x; o[5] = (bf16)b.y; o[6] = (bf16)b.z; o[7] = (bf16)b.w;
    *(v8bf*)(xb + (long)i * 8) = o;
}

// ---------------- Kernel 0b: W (C x H f32) -> B-operand-swizzled bf16 -------
__global__ void swz_w(const float* __restrict__ w, bf16* __restrict__ wsw) {
    int tid = blockIdx.x * blockDim.x + threadIdx.x;
    if (tid >= (CC / 32) * 4 * 32) return;
    int ks   = tid >> 7;
    int nt   = (tid >> 5) & 3;
    int lane = tid & 31;
    int c0 = ks * 32 + ((lane & 16) ? 8 : 0);
    int h  = nt * 16 + (lane & 15);
    bf16* dst = wsw + (long)tid * 16;
#pragma unroll
    for (int j = 0; j < 8; ++j) dst[j]     = (bf16)w[(c0 + j) * HD + h];
#pragma unroll
    for (int j = 0; j < 8; ++j) dst[8 + j] = (bf16)w[(c0 + 16 + j) * HD + h];
}

// ---------------- Kernel 1: QKV projection, register double-buffered --------
__global__ void qkv_proj(const bf16* __restrict__ xb, const bf16* __restrict__ wsw,
                         bf16* __restrict__ qb, bf16* __restrict__ kb,
                         bf16* __restrict__ vt) {
    int lane = threadIdx.x & 31;
    int w    = threadIdx.x >> 5;     // 0..5
    int mat  = w % 3;                // 0=q, 1=k, 2=v
    int mrow = w / 3;                // 0..1
    long m0 = (long)blockIdx.x * 32 + mrow * 16;   // global row (b*T + t)

    const bf16* wb    = wsw + (long)mat * (CC * HD);
    const bf16* abase = xb + m0 * CC;

    v8f acc[4];
#pragma unroll
    for (int nt = 0; nt < 4; ++nt) acc[nt] = vzero();

    // Double-buffered operands: prefetch step kk+1 while WMMAs of kk run,
    // so the compiler can issue partial s_wait_loadcnt instead of draining.
    v16bf a[2];
    v16bf bm[2][4];

    a[0] = load_ab16(abase, CC, lane);
#pragma unroll
    for (int nt = 0; nt < 4; ++nt) {
        const bf16* bbp = wb + ((long)nt * 32 + lane) * 16;
        union { v16bf v; v8bf h[2]; } bu;
        bu.h[0] = *(const v8bf*)(bbp);
        bu.h[1] = *(const v8bf*)(bbp + 8);
        bm[0][nt] = bu.v;
    }

#pragma unroll 2
    for (int kk = 0; kk < CC / 32; ++kk) {
        int cur = kk & 1, nxt = cur ^ 1;
        int kn = (kk + 1 < CC / 32) ? kk + 1 : kk;   // last iter: harmless reload
        a[nxt] = load_ab16(abase + kn * 32, CC, lane);
#pragma unroll
        for (int nt = 0; nt < 4; ++nt) {
            const bf16* bbp = wb + ((long)(kn * 4 + nt) * 32 + lane) * 16;
            union { v16bf v; v8bf h[2]; } bu;
            bu.h[0] = *(const v8bf*)(bbp);
            bu.h[1] = *(const v8bf*)(bbp + 8);
            bm[nxt][nt] = bu.v;
        }
#pragma unroll
        for (int nt = 0; nt < 4; ++nt)
            acc[nt] = wmma_bf16(a[cur], bm[cur][nt], acc[nt]);
    }

    int col    = lane & 15;
    int rowoff = (lane & 16) ? 8 : 0;
    if (mat == 2) {
        // V -> Vt[b][h][t] (h-major) so PV's B-operand source is N-major.
        long b  = m0 / TT;
        long t0 = (m0 % TT) + rowoff;
#pragma unroll
        for (int nt = 0; nt < 4; ++nt) {
            v8bf pk;
#pragma unroll
            for (int r = 0; r < 8; ++r) pk[r] = (bf16)acc[nt][r];
            *(v8bf*)(vt + ((b * HD) + nt * 16 + col) * (long)TT + t0) = pk;
        }
    } else {
        bf16* dst = (mat == 0) ? qb : kb;   // row-major (b*T + t, h)
#pragma unroll
        for (int nt = 0; nt < 4; ++nt)
#pragma unroll
            for (int r = 0; r < 8; ++r)
                dst[(m0 + rowoff + r) * HD + nt * 16 + col] = (bf16)acc[nt][r];
    }
}

// ---------------- Kernel 2: flash attention with async-LDS K/V staging ------
// 8 waves/block, all in the SAME batch -> K/V chunks are shared: stage them
// once per block via global_load_async_to_lds_b128, double-buffered.
__global__ void attn(const bf16* __restrict__ qb, const bf16* __restrict__ kb,
                     const bf16* __restrict__ vt, const int* __restrict__ mask,
                     float* __restrict__ out) {
    __shared__ __align__(16) bf16 kbuf[2][32][KPAD];  // 32 keys x 64 h (+pad)
    __shared__ __align__(16) bf16 vbuf[2][HD][VPAD];  // 64 h x 32 keys (+pad)
    __shared__ __align__(16) bf16 plds[8][16][PPAD];  // per-wave P staging

    int tid  = threadIdx.x;
    int lane = tid & 31;
    int w    = tid >> 5;
    int g    = blockIdx.x * 8 + w;          // global 16-query tile id
    int b    = g >> 7;                      // 128 tiles per batch
    int qt   = g & 127;
    long q0  = (long)b * TT + qt * 16;

    const float kLogScale = 0.125f * 1.44269504088896340736f;  // H^-0.5 * log2(e)

    v16bf aq0 = load_ab16(qb + q0 * HD,      HD, lane);
    v16bf aq1 = load_ab16(qb + q0 * HD + 32, HD, lane);

    v8f o[4];
#pragma unroll
    for (int nt = 0; nt < 4; ++nt) o[nt] = vzero();
    float mrow[8], lrow[8];
#pragma unroll
    for (int r = 0; r < 8; ++r) { mrow[r] = -1e30f; lrow[r] = 0.0f; }

    int col    = lane & 15;
    int rowoff = (lane & 16) ? 8 : 0;

    // Cooperative async staging of one 32-key chunk: 256 threads x 2 x b128.
    const bf16* kbb = kb + (long)b * TT * HD;
    const bf16* vbb = vt + (long)b * HD * TT;
    int krow = tid >> 3, kseg = tid & 7;    // K: 32 rows x 128B
    int vrow = tid >> 2, vseg = tid & 3;    // V: 64 rows x 64B

#define ISSUE_CHUNK(buf, K0)                                                   \
    do {                                                                       \
        async_b128(&kbuf[buf][krow][kseg * 8],                                 \
                   kbb + (long)((K0) + krow) * HD + kseg * 8);                 \
        async_b128(&vbuf[buf][vrow][vseg * 8],                                 \
                   vbb + (long)vrow * TT + (K0) + vseg * 8);                   \
    } while (0)

    ISSUE_CHUNK(0, 0);
    wait_async0();
    __syncthreads();

    for (int kc = 0; kc < TT / 32; ++kc) {
        int k0  = kc * 32;
        int cur = kc & 1;
        if (kc + 1 < TT / 32) ISSUE_CHUNK(cur ^ 1, k0 + 32);

        // S = Q * K^T from LDS-staged K (row-major: key x h)
        v8f s[2];
#pragma unroll
        for (int t = 0; t < 2; ++t) {
            v16bf bk0 = load_ab16(&kbuf[cur][t * 16][0],  KPAD, lane);
            v16bf bk1 = load_ab16(&kbuf[cur][t * 16][32], KPAD, lane);
            v8f z = vzero();
            z = wmma_bf16(aq0, bk0, z);
            z = wmma_bf16(aq1, bk1, z);
            float bias = mask[b * TT + k0 + t * 16 + col] ? 0.0f : -1e30f;
#pragma unroll
            for (int r = 0; r < 8; ++r) s[t][r] = z[r] * kLogScale + bias;
        }

        // Online softmax (rows match WMMA C-layout halves).
        float al[8];
#pragma unroll
        for (int r = 0; r < 8; ++r) {
            float m = fmaxf(s[0][r], s[1][r]);
#pragma unroll
            for (int off = 1; off < 16; off <<= 1)
                m = fmaxf(m, __shfl_xor(m, off, 32));
            float mnew = fmaxf(mrow[r], m);
            al[r]   = __builtin_amdgcn_exp2f(mrow[r] - mnew);
            mrow[r] = mnew;
            float p0 = __builtin_amdgcn_exp2f(s[0][r] - mnew);
            float p1 = __builtin_amdgcn_exp2f(s[1][r] - mnew);
            int row = r + rowoff;
            plds[w][row][col]      = (bf16)p0;
            plds[w][row][col + 16] = (bf16)p1;
            float ps = p0 + p1;
#pragma unroll
            for (int off = 1; off < 16; off <<= 1)
                ps += __shfl_xor(ps, off, 32);
            lrow[r] = lrow[r] * al[r] + ps;
        }
        __syncthreads();

        // Reload P as A-operand (16x32) from LDS.
        union { v16bf v; v8bf h[2]; } ap;
        {
            int prow = lane & 15;
            int pk   = (lane & 16) ? 8 : 0;
            ap.h[0] = *(const v8bf*)&plds[w][prow][pk];
            ap.h[1] = *(const v8bf*)&plds[w][prow][pk + 16];
        }

        // O = diag(alpha)*O + P*V from LDS-staged Vt (h-major: h x key)
#pragma unroll
        for (int nt = 0; nt < 4; ++nt) {
#pragma unroll
            for (int r = 0; r < 8; ++r) o[nt][r] *= al[r];
            v16bf bv = load_ab16(&vbuf[cur][nt * 16][0], VPAD, lane);
            o[nt] = wmma_bf16(ap.v, bv, o[nt]);
        }

        wait_async0();        // next chunk resident before buffer swap
        __syncthreads();
    }
#undef ISSUE_CHUNK

    // Epilogue: normalize and store f32 (coalesced 16-float row segments).
#pragma unroll
    for (int nt = 0; nt < 4; ++nt)
#pragma unroll
        for (int r = 0; r < 8; ++r) {
            int row = r + rowoff;
            out[(q0 + row) * HD + nt * 16 + col] = o[nt][r] / lrow[r];
        }
}

// ---------------- Host launcher ---------------------------------------------
extern "C" void kernel_launch(void* const* d_in, const int* in_sizes, int n_in,
                              void* d_out, int out_size, void* d_ws, size_t ws_size,
                              hipStream_t stream) {
    const float* x    = (const float*)d_in[0];
    const int*   amsk = (const int*)d_in[1];
    const float* Wk   = (const float*)d_in[2];
    const float* Wq   = (const float*)d_in[3];
    const float* Wv   = (const float*)d_in[4];
    float* out = (float*)d_out;

    // Workspace: xb 32MB | qb 2MB | kb 2MB | vt 2MB | wsw 384KB
    char* ws = (char*)d_ws;
    const size_t XB = (size_t)BB * TT * CC * 2;
    const size_t QK = (size_t)BB * TT * HD * 2;
    bf16* xb  = (bf16*)(ws);
    bf16* qb  = (bf16*)(ws + XB);
    bf16* kb  = (bf16*)(ws + XB + QK);
    bf16* vt  = (bf16*)(ws + XB + 2 * QK);
    bf16* wsw = (bf16*)(ws + XB + 3 * QK);

    int n8 = (BB * TT * CC) / 8;
    hipLaunchKernelGGL(cvt_x, dim3((n8 + 255) / 256), dim3(256), 0, stream, x, xb, n8);
    hipLaunchKernelGGL(swz_w, dim3(16), dim3(256), 0, stream, Wq, wsw + 0 * (CC * HD));
    hipLaunchKernelGGL(swz_w, dim3(16), dim3(256), 0, stream, Wk, wsw + 1 * (CC * HD));
    hipLaunchKernelGGL(swz_w, dim3(16), dim3(256), 0, stream, Wv, wsw + 2 * (CC * HD));
    hipLaunchKernelGGL(qkv_proj, dim3((BB * TT) / 32), dim3(192), 0, stream,
                       xb, wsw, qb, kb, vt);
    hipLaunchKernelGGL(attn, dim3((BB * TT) / 16 / 8), dim3(256), 0, stream,
                       qb, kb, vt, amsk, out);
}
